// GatedGraphConvolution_62637803045231
// MI455X (gfx1250) — matrix operand order repaired and checked by reading, not compiled, not run on hardware
//
#include <hip/hip_runtime.h>
#include <hip/hip_bf16.h>

typedef __attribute__((ext_vector_type(16))) _Float16 v16h;
typedef __attribute__((ext_vector_type(8)))  float    v8f;
typedef _Float16 half_t;

#define ND 64
#define INF 192
#define KC1 32
#define KC2 27

// ---------------------------------------------------------------- copy kernel
__global__ void ggc_copy_kernel(const float4* __restrict__ in,
                                float4* __restrict__ out, int n4) {
    int i = blockIdx.x * blockDim.x + threadIdx.x;
    if (i < n4) out[i] = in[i];
}

// ---------------------------------------------------------------- edge kernel
// 256 threads = 8 wave32; each wave processes a tile of 16 edges via WMMA.
__launch_bounds__(256, 1)
__global__ void ggc_edge_kernel(const float* __restrict__ input,
                                const int*  __restrict__ esrc,
                                const int*  __restrict__ etgt,
                                const float* __restrict__ rij,
                                const float* __restrict__ cs,
                                const float* __restrict__ pw,
                                const float* __restrict__ cutp,
                                const float* __restrict__ Wg, const float* __restrict__ bg,
                                const float* __restrict__ Wm, const float* __restrict__ bm,
                                const float* __restrict__ W1, const float* __restrict__ b1,
                                const float* __restrict__ W2g, const float* __restrict__ b2g,
                                const float* __restrict__ W2, const float* __restrict__ b2,
                                float* __restrict__ out, int nEdges)
{
    // Weights transposed to [col][k] in f16 so a lane's B-fragment
    // (k = 8*hi+0..7 and 16+8*hi+0..7 of a 32-chunk, via the kp swizzle) is one 32B contiguous run.
    __shared__ __align__(32) half_t sWg[ND * INF];   // 24 KB
    __shared__ __align__(32) half_t sWm[ND * INF];   // 24 KB
    __shared__ __align__(32) half_t sW1[ND * KC1];   // 4 KB
    __shared__ __align__(32) half_t sW2[ND * KC1];   // 4 KB (rows 27..31 zero)
    __shared__ float sW2g[KC2 * KC2];
    __shared__ float sb2g[KC2];
    __shared__ float sbg[ND], sbm[ND], sbz[ND];

    const int tid = threadIdx.x;

    // ---- cooperative weight preload (reads coalesced, writes transposed)
    for (int i = tid; i < INF * ND; i += 256) {
        int k = i >> 6, col = i & 63;                 // Wg/Wm are [192][64]
        int kp = k ^ ((((k >> 3) ^ (k >> 4)) & 1) * 24);
        sWg[col * INF + kp] = (half_t)Wg[i];
        sWm[col * INF + kp] = (half_t)Wm[i];
    }
    for (int i = tid; i < KC1 * ND; i += 256) {
        int k = i >> 6, col = i & 63;                 // W1 is [32][64]
        int kp = k ^ ((((k >> 3) ^ (k >> 4)) & 1) * 24);
        sW1[col * KC1 + kp] = (half_t)W1[i];
        float w2v = (k < KC2) ? W2[k * ND + col] : 0.0f;  // W2 is [27][64], pad K
        sW2[col * KC1 + kp] = (half_t)w2v;
    }
    for (int i = tid; i < KC2 * KC2; i += 256) sW2g[i] = W2g[i];
    if (tid < KC2) sb2g[tid] = b2g[tid];
    if (tid < ND)  { sbg[tid] = bg[tid]; sbm[tid] = bm[tid]; sbz[tid] = b1[tid] + b2[tid]; }
    __syncthreads();

    const int wave = tid >> 5;
    const int lane = tid & 31;
    const int tile = blockIdx.x * 8 + wave;
    if (tile * 16 >= nEdges) return;

    const int e0  = tile * 16;
    const int m   = lane & 15;       // A-matrix row (edge within tile)
    const int hi  = lane >> 4;       // lane half selects K sub-runs
    const int col = lane & 15;       // B/C/D column within 16-wide tile
    const int e   = e0 + m;

    const float cutoff = cutp[0];
    const float rinv   = 1.0f / rij[e];

    // ---- gather ni / nj rows as 4 runs of 8 floats matching A-frag K pattern:
    //      chunk c run r -> k = 32c + 16r + 8*hi + (0..7)
    const float4* nip = (const float4*)(input + (size_t)esrc[e] * ND + hi * 8);
    const float4* njp = (const float4*)(input + (size_t)etgt[e] * ND + hi * 8);
    float4 niA[8], njA[8];
#pragma unroll
    for (int r = 0; r < 4; ++r) {
        niA[2*r]   = nip[r*4];  niA[2*r+1] = nip[r*4+1];
        njA[2*r]   = njp[r*4];  njA[2*r+1] = njp[r*4+1];
    }

    // ---- build f16 A fragments: fe = [ni | nj | (ni-nj)/r], 6 K-chunks of 32
    v16h aN[2], aJ[2], aD[2];
#pragma unroll
    for (int c = 0; c < 2; ++c) {
        const float* pn = (const float*)&niA[c*4];
        const float* pj = (const float*)&njA[c*4];
#pragma unroll
        for (int x = 0; x < 16; ++x) {
            float vn = pn[x], vj = pj[x];
            aN[c][x] = (_Float16)vn;
            aJ[c][x] = (_Float16)vj;
            aD[c][x] = (_Float16)((vn - vj) * rinv);
        }
    }

    // ---- accumulators init with biases (bias depends on column only)
    v8f accG[4], accM[4], accZ[4];
#pragma unroll
    for (int n = 0; n < 4; ++n) {
        float vg = sbg[n*16 + col], vm = sbm[n*16 + col], vz = sbz[n*16 + col];
#pragma unroll
        for (int j = 0; j < 8; ++j) { accG[n][j] = vg; accM[n][j] = vm; accZ[n][j] = vz; }
    }

    // ---- main GEMMs: e_gate_pre, e_mlp_pre  (16x192 @ 192x64)
    const int kb = 16 * hi;
#pragma unroll
    for (int n = 0; n < 4; ++n) {
        const half_t* wgb = &sWg[(n*16 + col) * INF + kb];
        const half_t* wmb = &sWm[(n*16 + col) * INF + kb];
#pragma unroll
        for (int c = 0; c < 6; ++c) {
            v16h bG = *(const v16h*)(wgb + c * 32);
            v16h bM = *(const v16h*)(wmb + c * 32);
            v16h a  = (c < 2) ? aN[c] : (c < 4) ? aJ[c-2] : aD[c-4];
            accG[n] = __builtin_amdgcn_wmma_f32_16x16x32_f16(false, a, false, bG,
                                                             (short)0, accG[n], false, false);
            accM[n] = __builtin_amdgcn_wmma_f32_16x16x32_f16(false, a, false, bM,
                                                             (short)0, accM[n], false, false);
        }
    }

    // ---- combine_sets A fragment (16x32)
    const float* csrow = cs + (size_t)e * KC1 + hi * 8;
    float4 c0 = *(const float4*)(csrow);
    float4 c1 = *(const float4*)(csrow + 4);
    float4 c2 = *(const float4*)(csrow + 16);
    float4 c3 = *(const float4*)(csrow + 20);
    v16h aC;
    {
        const float* p0 = (const float*)&c0;
#pragma unroll
        for (int x = 0; x < 8;  ++x) aC[x] = (_Float16)p0[x];   // c0,c1 contiguous
        const float* p2 = (const float*)&c2;
#pragma unroll
        for (int x = 0; x < 8;  ++x) aC[8 + x] = (_Float16)p2[x]; // c2,c3 contiguous
    }

    // ---- plane_wave gated A fragment: (pw * sigmoid(pw@W2g + b2g)), K padded to 32
    const float* pwrow = pw + (size_t)e * KC2;
    float pr[KC2];
#pragma unroll
    for (int j = 0; j < KC2; ++j) pr[j] = pwrow[j];
    v16h aP;
#pragma unroll
    for (int x = 0; x < 16; ++x) {
        int kk = (x < 8 ? x : x + 8) + hi * 8;
        float v = 0.0f;
        if (kk < KC2) {
            float g = sb2g[kk];
#pragma unroll
            for (int j = 0; j < KC2; ++j) g += pr[j] * sW2g[j * KC2 + kk];
            g = 1.0f / (1.0f + __expf(-g));
            v = pwrow[kk] * g;
        }
        aP[x] = (_Float16)v;
    }

    // ---- z = combine_sets@W1 + pw_gated@W2  (+ b1 + b2 already in accZ)
#pragma unroll
    for (int n = 0; n < 4; ++n) {
        v16h b1f = *(const v16h*)(&sW1[(n*16 + col) * KC1 + kb]);
        v16h b2f = *(const v16h*)(&sW2[(n*16 + col) * KC1 + kb]);
        accZ[n] = __builtin_amdgcn_wmma_f32_16x16x32_f16(false, aC, false, b1f,
                                                         (short)0, accZ[n], false, false);
        accZ[n] = __builtin_amdgcn_wmma_f32_16x16x32_f16(false, aP, false, b2f,
                                                         (short)0, accZ[n], false, false);
    }

    // ---- epilogue: sigmoid(g) * softplus(m) * z * mask, scatter-add to out
    // C/D layout: VGPR j holds row (j + 8*hi), column (n*16 + col)
    int   rowsrc [8];
    float rowmask[8];
#pragma unroll
    for (int j = 0; j < 8; ++j) {
        int ee = e0 + j + 8 * hi;
        rowsrc[j]  = esrc[ee];
        rowmask[j] = (rij[ee] < cutoff) ? 1.0f : 0.0f;
    }
#pragma unroll
    for (int n = 0; n < 4; ++n) {
        int cc = n * 16 + col;
#pragma unroll
        for (int j = 0; j < 8; ++j) {
            float g  = accG[n][j];
            float mm = accM[n][j];
            float zz = accZ[n][j];
            float sg = 1.0f / (1.0f + __expf(-g));
            float sp = (mm > 20.0f) ? mm : __logf(1.0f + __expf(mm));
            float val = sg * sp * zz * rowmask[j];
            atomicAdd(out + (size_t)rowsrc[j] * ND + cc, val);
        }
    }
}

// ---------------------------------------------------------------- launcher
extern "C" void kernel_launch(void* const* d_in, const int* in_sizes, int n_in,
                              void* d_out, int out_size, void* d_ws, size_t ws_size,
                              hipStream_t stream) {
    const float* input = (const float*)d_in[0];
    const int*   esrc  = (const int*)  d_in[1];
    const int*   etgt  = (const int*)  d_in[2];
    const float* rij   = (const float*)d_in[3];
    const float* cs    = (const float*)d_in[4];
    const float* pw    = (const float*)d_in[5];
    const float* cutp  = (const float*)d_in[6];
    const float* Wg    = (const float*)d_in[7];
    const float* bg    = (const float*)d_in[8];
    const float* Wm    = (const float*)d_in[9];
    const float* bm    = (const float*)d_in[10];
    const float* W1    = (const float*)d_in[11];
    const float* b1    = (const float*)d_in[12];
    const float* W2g   = (const float*)d_in[13];
    const float* b2g   = (const float*)d_in[14];
    const float* W2    = (const float*)d_in[15];
    const float* b2    = (const float*)d_in[16];
    float* out = (float*)d_out;

    const int nEdges = in_sizes[3];
    const int n4     = out_size / 4;          // out = [N_NODES, 64] f32

    // out = input
    ggc_copy_kernel<<<(n4 + 255) / 256, 256, 0, stream>>>(
        (const float4*)input, (float4*)out, n4);

    // scatter-add gated messages
    const int tiles  = (nEdges + 15) / 16;
    const int blocks = (tiles + 7) / 8;
    ggc_edge_kernel<<<blocks, 256, 0, stream>>>(
        input, esrc, etgt, rij, cs, pw, cutp,
        Wg, bg, Wm, bm, W1, b1, W2g, b2g, W2, b2,
        out, nEdges);
}
